// CGCNNModel_49194555408406
// MI455X (gfx1250) — matrix operand order repaired and checked by reading, not compiled, not run on hardware
//
#include <hip/hip_runtime.h>
#include <hip/hip_bf16.h>

// ---------------------------------------------------------------------------
// CGCNN-style GNN forward for MI455X (gfx1250, wave32).
// Node-feature GEMMs on v_wmma_f32_16x16x32_f16 with direct per-lane b128
// fragment loads from fp16-padded operands (no guards, no LDS staging).
// Edge softmax / message scatter use float atomics (memory-bound path).
// ---------------------------------------------------------------------------

typedef __attribute__((ext_vector_type(16))) _Float16 v16h;
typedef __attribute__((ext_vector_type(8)))  _Float16 v8h;
typedef __attribute__((ext_vector_type(8)))  float    v8f;

#define SG    64          // graphs
#define NPG   558         // nodes per graph
#define NN    35712       // SG*NPG
#define E0    384000      // raw edges (SG*6000)
#define EL    419712      // E0 + NN self loops
#define NH    4           // heads
#define HD    64          // head dim
#define C256  256
#define CCAT  328

// ------------------------------- utilities --------------------------------

__device__ inline void atomicMaxF(float* addr, float v) {
  int* ai = (int*)addr;
  int old = __float_as_int(*addr);
  while (v > __int_as_float(old)) {
    int assumed = old;
    old = atomicCAS(ai, assumed, __float_as_int(v));
    if (old == assumed) break;
  }
}

__global__ __launch_bounds__(256) void fill_kernel(float* p, size_t n, float v) {
  size_t i = (size_t)blockIdx.x * blockDim.x + threadIdx.x;
  if (i < n) p[i] = v;
}

// -------------------------- edge preprocessing -----------------------------

__global__ __launch_bounds__(256) void bd_max_kernel(const float* bd, float* bdmax) {
  int i = blockIdx.x * blockDim.x + threadIdx.x;
  if (i < E0) atomicMaxF(bdmax, bd[i]);
}

__global__ __launch_bounds__(256) void build_edges_kernel(
    const int* ei, const float* bd, const float* bdmax,
    int* srcL, int* dstL, float* ww) {
  int i = blockIdx.x * blockDim.x + threadIdx.x;
  if (i >= EL) return;
  if (i < E0) {
    int g = i / 6000, j = i % 6000;
    srcL[i] = ei[g * 12000 + j] + g * NPG;
    dstL[i] = ei[g * 12000 + 6000 + j] + g * NPG;
    ww[i]   = bd[i] / bdmax[0];
  } else {
    int n = i - E0;
    srcL[i] = n; dstL[i] = n; ww[i] = 1.0f;
  }
}

// ----------------------------- projections ---------------------------------

__global__ __launch_bounds__(256) void proj_sg_kernel(
    const float* xim, const float* xsg, const float* W, const float* b, float* out) {
  int i = blockIdx.x * blockDim.x + threadIdx.x;
  if (i >= NN * 32) return;
  int n = i / 32, o = i % 32;
  float s = b[o];
  #pragma unroll
  for (int k = 0; k < 8; ++k) s += xim[n * 8 + k] * W[k * 32 + o];
  #pragma unroll
  for (int k = 0; k < 8; ++k) s += xsg[n * 8 + k] * W[(8 + k) * 32 + o];
  out[i] = fmaxf(s, 0.0f);
}

__global__ __launch_bounds__(256) void dense_kernel(
    const float* X, const float* W, const float* b, float* Y,
    int M, int K, int O, int relu) {
  int i = blockIdx.x * blockDim.x + threadIdx.x;
  if (i >= M * O) return;
  int m = i / O, o = i % O;
  float s = b[o];
  for (int k = 0; k < K; ++k) s += X[m * K + k] * W[k * O + o];
  Y[i] = relu ? fmaxf(s, 0.0f) : s;
}

// --------------------- fp16 operand preparation ----------------------------
// Ah[n, k] = fp16(A[n, k]) with zero pad k in [K, Kp); Kp % 32 == 0.

__global__ __launch_bounds__(256) void cvt_a_kernel(
    const float* __restrict__ A, int lda, int K, int Kp, _Float16* __restrict__ Ah) {
  size_t i = (size_t)blockIdx.x * blockDim.x + threadIdx.x;
  if (i >= (size_t)NN * Kp) return;
  int n = (int)(i / Kp), k = (int)(i % Kp);
  Ah[i] = (_Float16)((k < K) ? A[(size_t)n * lda + k] : 0.0f);
}

// Wt[n, k] = fp16(W[k, n])  (N-major so B-fragment K runs are contiguous)
__global__ __launch_bounds__(256) void cvt_wt_kernel(
    const float* __restrict__ W, int K, int Kp, _Float16* __restrict__ Wt) {
  int i = blockIdx.x * blockDim.x + threadIdx.x;
  if (i >= C256 * Kp) return;
  int n = i / Kp, k = i % Kp;
  Wt[i] = (_Float16)((k < K) ? W[(size_t)k * C256 + n] : 0.0f);
}

// --------------------------- WMMA GEMM (N=256) -----------------------------
// C[M,256] = Ah[M,KP] * Wt[256,KP]^T ; block = 256 thr = 8 waves.
// Wave w: rows m0 = (blockIdx.x*8+w)*16, cols n0 = blockIdx.y*64 (4 N-tiles).
// Fragment loads are contiguous 16B per lane (global_load_b128), fully
// unrolled over KP (template) -> dense wmma + vmem mix, no branches.

template <int KP>
__global__ __launch_bounds__(256) void gemm_wmma_kernel(
    const _Float16* __restrict__ Ah,
    const _Float16* __restrict__ Wt,
    float* __restrict__ C, int ldc) {
  const int wave = threadIdx.x >> 5;
  const int lane = threadIdx.x & 31;
  const int m0 = (blockIdx.x * 8 + wave) << 4;
  const int n0 = blockIdx.y << 6;
  const int nlo  = lane & 15;
  const int khiA = (lane & 16) ? 16 : 0;   // A hi lane-half: K chunk +16
  const int khiB = (lane & 16) ? 16 : 0;   // B hi lane-half: K chunk +16

  const _Float16* ap  = Ah + (size_t)(m0 + nlo) * KP + khiA;
  const _Float16* bp0 = Wt + (size_t)(n0 + nlo) * KP + khiB;
  const _Float16* bp1 = bp0 + (size_t)16 * KP;
  const _Float16* bp2 = bp0 + (size_t)32 * KP;
  const _Float16* bp3 = bp0 + (size_t)48 * KP;

  v8f acc0 = {}, acc1 = {}, acc2 = {}, acc3 = {};

  #pragma unroll
  for (int k0 = 0; k0 < KP; k0 += 32) {
    union { v16h v; v8h h[2]; } a, b0, b1, b2, b3;
    a.h[0]  = *(const v8h*)(ap + k0);        // K = khiA + 0..7
    a.h[1]  = *(const v8h*)(ap + k0 + 8);    // K = khiA + 8..15
    b0.h[0] = *(const v8h*)(bp0 + k0);       // K = khiB + 0..7
    b0.h[1] = *(const v8h*)(bp0 + k0 + 8);   // K = khiB + 8..15
    b1.h[0] = *(const v8h*)(bp1 + k0);
    b1.h[1] = *(const v8h*)(bp1 + k0 + 8);
    b2.h[0] = *(const v8h*)(bp2 + k0);
    b2.h[1] = *(const v8h*)(bp2 + k0 + 8);
    b3.h[0] = *(const v8h*)(bp3 + k0);
    b3.h[1] = *(const v8h*)(bp3 + k0 + 8);
    acc0 = __builtin_amdgcn_wmma_f32_16x16x32_f16(false, a.v, false, b0.v,
                                                  (short)0, acc0, false, false);
    acc1 = __builtin_amdgcn_wmma_f32_16x16x32_f16(false, a.v, false, b1.v,
                                                  (short)0, acc1, false, false);
    acc2 = __builtin_amdgcn_wmma_f32_16x16x32_f16(false, a.v, false, b2.v,
                                                  (short)0, acc2, false, false);
    acc3 = __builtin_amdgcn_wmma_f32_16x16x32_f16(false, a.v, false, b3.v,
                                                  (short)0, acc3, false, false);
  }

  // C/D layout: VGPR r -> row r (+8 for hi lane-half), col = lane&15
  const int rbase = m0 + ((lane & 16) ? 8 : 0);
  #pragma unroll
  for (int r = 0; r < 8; ++r) {
    size_t row = (size_t)(rbase + r) * ldc;
    C[row + n0 +      nlo] = acc0[r];
    C[row + n0 + 16 + nlo] = acc1[r];
    C[row + n0 + 32 + nlo] = acc2[r];
    C[row + n0 + 48 + nlo] = acc3[r];
  }
}

// ------------------------------- GAT pieces --------------------------------

__global__ __launch_bounds__(256) void att_scores_kernel(
    const float* h, const float* att_src, const float* att_dst,
    float* asrc, float* adst) {
  int i = blockIdx.x * blockDim.x + threadIdx.x;
  if (i >= NN * NH) return;
  int n = i / NH, hh = i % NH;
  const float* hp = h + (size_t)n * C256 + hh * HD;
  float ss = 0.f, sd = 0.f;
  #pragma unroll 8
  for (int d = 0; d < HD; ++d) {
    float v = hp[d];
    ss += v * att_src[hh * HD + d];
    sd += v * att_dst[hh * HD + d];
  }
  asrc[i] = ss; adst[i] = sd;
}

__global__ __launch_bounds__(256) void edge_logit_max_kernel(
    const int* srcL, const int* dstL, const float* asrc, const float* adst,
    float* alpha, float* segmax) {
  int i = blockIdx.x * blockDim.x + threadIdx.x;
  if (i >= EL * NH) return;
  int e = i / NH, hh = i % NH;
  int s = srcL[e], t = dstL[e];
  float l = asrc[s * NH + hh] + adst[t * NH + hh];
  l = (l > 0.f) ? l : 0.2f * l;              // leaky relu, slope 0.2
  alpha[i] = l;
  atomicMaxF(&segmax[t * NH + hh], l);
}

__global__ __launch_bounds__(256) void edge_exp_sum_kernel(
    const int* dstL, float* alpha, const float* segmax, float* segsum) {
  int i = blockIdx.x * blockDim.x + threadIdx.x;
  if (i >= EL * NH) return;
  int e = i / NH, hh = i % NH;
  int t = dstL[e];
  float ex = __expf(alpha[i] - segmax[t * NH + hh]);
  alpha[i] = ex;
  atomicAdd(&segsum[t * NH + hh], ex);
}

// one block per edge, 256 threads = 256 features (head = f>>6)
__global__ __launch_bounds__(256) void scatter_msg_kernel(
    const int* srcL, const int* dstL, const float* h,
    const float* alpha, const float* segsum, float* out) {
  int e = blockIdx.x, f = threadIdx.x;
  int hh = f >> 6;
  int s = srcL[e], t = dstL[e];
  float w = alpha[(size_t)e * NH + hh] / (segsum[t * NH + hh] + 1e-16f);
  atomicAdd(&out[(size_t)t * C256 + f], h[(size_t)s * C256 + f] * w);
}

// ------------------------------- GCN pieces --------------------------------

__global__ __launch_bounds__(256) void deg_kernel(
    const int* dstL, const float* ww, float* deg) {
  int i = blockIdx.x * blockDim.x + threadIdx.x;
  if (i < EL) atomicAdd(&deg[dstL[i]], ww[i]);
}

__global__ __launch_bounds__(256) void dinv_kernel(const float* deg, float* dinv) {
  int i = blockIdx.x * blockDim.x + threadIdx.x;
  if (i >= NN) return;
  float d = deg[i];
  dinv[i] = (d > 0.f) ? rsqrtf(d) : 0.f;
}

__global__ __launch_bounds__(256) void normw_kernel(
    const int* srcL, const int* dstL, const float* ww, const float* dinv,
    float* normw) {
  int i = blockIdx.x * blockDim.x + threadIdx.x;
  if (i < EL) normw[i] = dinv[srcL[i]] * ww[i] * dinv[dstL[i]];
}

__global__ __launch_bounds__(256) void scatter_gcn_kernel(
    const int* srcL, const int* dstL, const float* h,
    const float* normw, float* out) {
  int e = blockIdx.x, f = threadIdx.x;
  atomicAdd(&out[(size_t)dstL[e] * C256 + f],
            h[(size_t)srcL[e] * C256 + f] * normw[e]);
}

// ------------------------- BatchNorm (train) + ReLU ------------------------
// value = x + bias (layer bias folded in), then BN over the node axis.

__global__ __launch_bounds__(256) void bn_stats_kernel(
    const float* x, int ld, const float* bias, float* sums, float* sumsq) {
  int f = threadIdx.x;                     // 256 features
  float s = 0.f, sq = 0.f;
  for (int n = blockIdx.x; n < NN; n += gridDim.x) {
    float v = x[(size_t)n * ld + f] + bias[f];
    s += v; sq += v * v;
  }
  atomicAdd(&sums[f], s);
  atomicAdd(&sumsq[f], sq);
}

__global__ __launch_bounds__(256) void bn_apply_kernel(
    const float* x, int ldin, const float* bias,
    const float* sums, const float* sumsq, float* y, int ldout) {
  size_t i = (size_t)blockIdx.x * blockDim.x + threadIdx.x;
  if (i >= (size_t)NN * C256) return;
  int n = (int)(i >> 8), f = (int)(i & 255);
  const float inv = 1.0f / (float)NN;
  float m  = sums[f] * inv;
  float vr = sumsq[f] * inv - m * m;
  float v  = x[(size_t)n * ldin + f] + bias[f];
  y[(size_t)n * ldout + f] = fmaxf((v - m) * rsqrtf(vr + 1e-5f), 0.0f);
}

// ----------------------------- concat + pool -------------------------------

__global__ __launch_bounds__(256) void concat_fill_kernel(
    const float* sg2d, const float* tex, const float* pres, float* x) {
  int i = blockIdx.x * blockDim.x + threadIdx.x;
  if (i >= NN * 72) return;
  int n = i / 72, c = i % 72;
  float v;
  if (c < 32)      v = sg2d[(size_t)n * 32 + c];
  else if (c < 64) v = tex[(n / NPG) * 32 + (c - 32)];
  else             v = pres[(n / NPG) * 8 + (c - 64)];
  x[(size_t)n * CCAT + 256 + c] = v;
}

__global__ __launch_bounds__(256) void pool_kernel(const float* x, float* pooled) {
  int s = blockIdx.x, f = threadIdx.x;   // 64 blocks x 256 threads
  float acc = 0.f;
  for (int i = 0; i < NPG; ++i)
    acc += x[((size_t)s * NPG + i) * C256 + f];
  pooled[s * C256 + f] = acc;
}

// ------------------------------- launcher ----------------------------------

extern "C" void kernel_launch(void* const* d_in, const int* in_sizes, int n_in,
                              void* d_out, int out_size, void* d_ws, size_t ws_size,
                              hipStream_t stream) {
  // jax tree-flatten (alphabetical) order of setup_inputs dict:
  const float* bd        = (const float*)d_in[1];   // bond_dist [64,6000]
  const int*   ei        = (const int*)  d_in[2];   // edge_index [64,2,6000]
  const float* att0W     = (const float*)d_in[4];
  const float* att0_adst = (const float*)d_in[5];
  const float* att0_asrc = (const float*)d_in[6];
  const float* att0_b    = (const float*)d_in[7];
  const float* att1W     = (const float*)d_in[8];
  const float* att1_adst = (const float*)d_in[9];
  const float* att1_asrc = (const float*)d_in[10];
  const float* att1_b    = (const float*)d_in[11];
  const float* conv0W    = (const float*)d_in[12];
  const float* conv0b    = (const float*)d_in[13];
  const float* conv1W    = (const float*)d_in[14];
  const float* conv1b    = (const float*)d_in[15];
  const float* fcW       = (const float*)d_in[16];
  const float* fcb       = (const float*)d_in[17];
  const float* hid0W     = (const float*)d_in[18];
  const float* hid0b     = (const float*)d_in[19];
  const float* hid1W     = (const float*)d_in[20];
  const float* hid1b     = (const float*)d_in[21];
  const float* presW     = (const float*)d_in[22];
  const float* presb     = (const float*)d_in[23];
  const float* sgW       = (const float*)d_in[24];
  const float* sgb       = (const float*)d_in[25];
  const float* texW      = (const float*)d_in[26];
  const float* texb      = (const float*)d_in[27];
  const float* xIM       = (const float*)d_in[28];
  const float* xnode     = (const float*)d_in[29];
  const float* xpres     = (const float*)d_in[30];
  const float* xsg       = (const float*)d_in[31];
  const float* xtex      = (const float*)d_in[32];
  float* out = (float*)d_out;

  // workspace carve-up (deterministic every call)
  char* ws = (char*)d_ws;
  size_t off = 0;
  auto alloc = [&](size_t bytes) -> char* {
    char* p = ws + off;
    off = (off + bytes + 255) & ~(size_t)255;
    return p;
  };
  int*      srcL   = (int*)     alloc((size_t)EL * 4);
  int*      dstL   = (int*)     alloc((size_t)EL * 4);
  float*    ww     = (float*)   alloc((size_t)EL * 4);
  float*    normw  = (float*)   alloc((size_t)EL * 4);
  float*    alpha  = (float*)   alloc((size_t)EL * NH * 4);
  float*    segmax = (float*)   alloc((size_t)NN * NH * 4);
  float*    segsum = (float*)   alloc((size_t)NN * NH * 4);
  float*    asrc   = (float*)   alloc((size_t)NN * NH * 4);
  float*    adst   = (float*)   alloc((size_t)NN * NH * 4);
  float*    deg    = (float*)   alloc((size_t)NN * 4);
  float*    dinv   = (float*)   alloc((size_t)NN * 4);
  float*    sums   = (float*)   alloc(512 * 4);
  float*    sumsq  = (float*)   alloc(512 * 4);
  float*    bdmax  = (float*)   alloc(256);
  float*    sg2d   = (float*)   alloc((size_t)NN * 32 * 4);
  float*    texb2  = (float*)   alloc((size_t)SG * 32 * 4);
  float*    presb2 = (float*)   alloc((size_t)SG * 8 * 4);
  float*    pooled = (float*)   alloc((size_t)SG * C256 * 4);
  float*    mlp1   = (float*)   alloc((size_t)SG * 128 * 4);
  float*    mlp2   = (float*)   alloc((size_t)SG * 64 * 4);
  _Float16* Ah     = (_Float16*)alloc((size_t)NN * 352 * 2);   // fp16 A, padded
  _Float16* Wth    = (_Float16*)alloc((size_t)352 * C256 * 2); // fp16 W^T, padded
  float*    bufA   = (float*)   alloc((size_t)NN * CCAT * 4);  // layer in / BN out
  float*    bufB   = (float*)   alloc((size_t)NN * C256 * 4);  // h = x @ W
  float*    bufC   = (float*)   alloc((size_t)NN * C256 * 4);  // aggregated msgs

  auto cdiv = [](size_t a, size_t b) { return (unsigned)((a + b - 1) / b); };
  auto fill = [&](float* p, size_t n, float v) {
    fill_kernel<<<cdiv(n, 256), 256, 0, stream>>>(p, n, v);
  };

  // GEMM wrapper: fp16 convert/pad/transpose, then templated WMMA kernel.
  auto gemm = [&](const float* A, int lda, int K, const float* W, float* C) {
    int Kp = (K + 31) & ~31;
    cvt_a_kernel<<<cdiv((size_t)NN * Kp, 256), 256, 0, stream>>>(A, lda, K, Kp, Ah);
    cvt_wt_kernel<<<cdiv((size_t)C256 * Kp, 256), 256, 0, stream>>>(W, K, Kp, Wth);
    dim3 g(NN / 16 / 8, C256 / 64);   // (279, 4)
    if (Kp == 96)
      gemm_wmma_kernel<96><<<g, 256, 0, stream>>>(Ah, Wth, C, C256);
    else if (Kp == 256)
      gemm_wmma_kernel<256><<<g, 256, 0, stream>>>(Ah, Wth, C, C256);
    else
      gemm_wmma_kernel<352><<<g, 256, 0, stream>>>(Ah, Wth, C, C256);
  };

  // --- edges, bond-dist normalization, projections ---
  fill(bdmax, 1, 0.0f);
  bd_max_kernel<<<cdiv(E0, 256), 256, 0, stream>>>(bd, bdmax);
  build_edges_kernel<<<cdiv(EL, 256), 256, 0, stream>>>(ei, bd, bdmax, srcL, dstL, ww);
  proj_sg_kernel<<<cdiv((size_t)NN * 32, 256), 256, 0, stream>>>(xIM, xsg, sgW, sgb, sg2d);
  dense_kernel<<<cdiv(SG * 32, 256), 256, 0, stream>>>(xtex, texW, texb, texb2, SG, 16, 32, 1);
  dense_kernel<<<cdiv(SG * 8, 256), 256, 0, stream>>>(xpres, presW, presb, presb2, SG, 4, 8, 1);

  // --- one GAT layer ---
  auto gat = [&](const float* X, int lda, int K, const float* W,
                 const float* aD, const float* aS, const float* bias,
                 float* ybuf, int ldy) {
    gemm(X, lda, K, W, bufB);
    att_scores_kernel<<<cdiv((size_t)NN * NH, 256), 256, 0, stream>>>(bufB, aS, aD, asrc, adst);
    fill(segmax, (size_t)NN * NH, -1e30f);
    fill(segsum, (size_t)NN * NH, 0.0f);
    edge_logit_max_kernel<<<cdiv((size_t)EL * NH, 256), 256, 0, stream>>>(
        srcL, dstL, asrc, adst, alpha, segmax);
    edge_exp_sum_kernel<<<cdiv((size_t)EL * NH, 256), 256, 0, stream>>>(
        dstL, alpha, segmax, segsum);
    fill(bufC, (size_t)NN * C256, 0.0f);
    scatter_msg_kernel<<<EL, 256, 0, stream>>>(srcL, dstL, bufB, alpha, segsum, bufC);
    fill(sums, 512, 0.0f); fill(sumsq, 512, 0.0f);
    bn_stats_kernel<<<256, 256, 0, stream>>>(bufC, C256, bias, sums, sumsq);
    bn_apply_kernel<<<cdiv((size_t)NN * C256, 256), 256, 0, stream>>>(
        bufC, C256, bias, sums, sumsq, ybuf, ldy);
  };

  // GAT1: x_node [NN,92] -> bufA [NN,256]
  gat(xnode, 92, 92, att0W, att0_adst, att0_asrc, att0_b, bufA, C256);
  // GAT2: bufA [NN,256] -> bufA as leading 256 cols of [NN,328]
  gat(bufA, C256, C256, att1W, att1_adst, att1_asrc, att1_b, bufA, CCAT);

  // concat global features into cols 256..327
  concat_fill_kernel<<<cdiv((size_t)NN * 72, 256), 256, 0, stream>>>(sg2d, texb2, presb2, bufA);

  // GCN normalization (shared by both layers)
  fill(deg, NN, 0.0f);
  deg_kernel<<<cdiv(EL, 256), 256, 0, stream>>>(dstL, ww, deg);
  dinv_kernel<<<cdiv(NN, 256), 256, 0, stream>>>(deg, dinv);
  normw_kernel<<<cdiv(EL, 256), 256, 0, stream>>>(srcL, dstL, ww, dinv, normw);

  // --- one GCN layer ---
  auto gcn = [&](const float* X, int lda, int K, const float* W, const float* bias,
                 float* ybuf, int ldy) {
    gemm(X, lda, K, W, bufB);
    fill(bufC, (size_t)NN * C256, 0.0f);
    scatter_gcn_kernel<<<EL, 256, 0, stream>>>(srcL, dstL, bufB, normw, bufC);
    fill(sums, 512, 0.0f); fill(sumsq, 512, 0.0f);
    bn_stats_kernel<<<256, 256, 0, stream>>>(bufC, C256, bias, sums, sumsq);
    bn_apply_kernel<<<cdiv((size_t)NN * C256, 256), 256, 0, stream>>>(
        bufC, C256, bias, sums, sumsq, ybuf, ldy);
  };

  gcn(bufA, CCAT, CCAT, conv0W, conv0b, bufA, C256);   // [NN,328] -> [NN,256]
  gcn(bufA, C256, C256, conv1W, conv1b, bufA, C256);   // [NN,256] -> [NN,256]

  // pool + MLP head
  pool_kernel<<<SG, 256, 0, stream>>>(bufA, pooled);
  dense_kernel<<<cdiv(SG * 128, 256), 256, 0, stream>>>(pooled, hid0W, hid0b, mlp1, SG, 256, 128, 1);
  dense_kernel<<<cdiv(SG * 64, 256), 256, 0, stream>>>(mlp1, hid1W, hid1b, mlp2, SG, 128, 64, 1);
  dense_kernel<<<cdiv(SG * 100, 256), 256, 0, stream>>>(mlp2, fcW, fcb, out, SG, 64, 100, 0);

  (void)in_sizes; (void)n_in; (void)out_size; (void)ws_size;
}
